// Attention_79998060855403
// MI455X (gfx1250) — matrix operand order repaired and checked by reading, not compiled, not run on hardware
//
#include <hip/hip_runtime.h>
#include <hip/hip_bf16.h>

typedef __attribute__((ext_vector_type(16))) __bf16 v16bf;
typedef __attribute__((ext_vector_type(8)))  __bf16 v8bf;
typedef __attribute__((ext_vector_type(4)))  __bf16 v4bf;
typedef __attribute__((ext_vector_type(2)))  __bf16 v2bf;
typedef __attribute__((ext_vector_type(8)))  float  v8f;

#define SEQ    2048
#define DMODEL 4096
#define NH     32
#define NKV    8
#define HD     128
#define KVCOLS (NKV * HD)   // 1024

union Frag16 { v16bf v; v8bf h[2]; };

__device__ inline v16bf mk_frag(const __bf16* p0, const __bf16* p1) {
    Frag16 f;
    f.h[0] = *(const v8bf*)p0;
    f.h[1] = *(const v8bf*)p1;
    return f.v;
}

// per-lane async copy of 16 bytes global -> LDS (ASYNCcnt-tracked)
__device__ inline void async_b128(unsigned lds_off, const void* gaddr) {
    asm volatile("global_load_async_to_lds_b128 %0, %1, off"
                 :: "v"(lds_off), "v"(gaddr) : "memory");
}
__device__ inline void wait_async0() {
    asm volatile("s_wait_asynccnt 0x0" ::: "memory");
}

// ---------------------------------------------------------------------------
// flat fp32 -> bf16 (count multiple of 1024)
// ---------------------------------------------------------------------------
__global__ __launch_bounds__(256)
void cvt_f32_bf16(const float* __restrict__ in, __bf16* __restrict__ out) {
    size_t i = ((size_t)blockIdx.x * 256 + threadIdx.x) * 4;
    float4 f = *(const float4*)&in[i];
    v4bf o;
    o[0] = (__bf16)f.x; o[1] = (__bf16)f.y; o[2] = (__bf16)f.z; o[3] = (__bf16)f.w;
    *(v4bf*)&out[i] = o;
}

// ---------------------------------------------------------------------------
// fp32 [R][C] -> bf16 [C][R], 32x32 LDS tiles (coalesced both sides)
// ---------------------------------------------------------------------------
__global__ __launch_bounds__(256)
void transpose_cvt(const float* __restrict__ in, __bf16* __restrict__ out,
                   int R, int C) {
    __shared__ float tile[32 * 33];
    const int tid = threadIdx.x;
    const int r0 = blockIdx.y * 32, c0 = blockIdx.x * 32;
#pragma unroll
    for (int i = 0; i < 4; ++i) {
        int e = tid + i * 256, tr = e >> 5, tc = e & 31;
        tile[tr * 33 + tc] = in[(size_t)(r0 + tr) * C + c0 + tc];
    }
    __syncthreads();
#pragma unroll
    for (int i = 0; i < 4; ++i) {
        int e = tid + i * 256, tr = e >> 5, tc = e & 31;
        out[(size_t)(c0 + tr) * R + r0 + tc] = (__bf16)tile[tc * 33 + tr];
    }
}

// ---------------------------------------------------------------------------
// C[M,N] = A[M,K] * Bt[N,K]^T.  bf16 in / fp32 out.
// BM=128 BN=128 BK=32; 8 waves x (32x64) = 2x4 WMMA tiles.
// Double-buffered LDS filled by global_load_async_to_lds_b128.
// ---------------------------------------------------------------------------
#define BM 128
#define BN 128
#define BK 32

__global__ __launch_bounds__(256)
void gemm_bf16_wmma(const __bf16* __restrict__ A, const __bf16* __restrict__ Bt,
                    float* __restrict__ C, int M, int N, int K) {
    __shared__ __bf16 sA[2][BM * BK];   // row-major [128][32]
    __shared__ __bf16 sB[2][BN * BK];   // n-major   [128][32]

    const int tid  = threadIdx.x;
    const int lane = tid & 31;
    const int wave = tid >> 5;
    const int wm   = wave & 3;          // 4 row-blocks of 32
    const int wn   = wave >> 2;         // 2 col-blocks of 64
    const int l16  = lane & 15;
    const int half = lane >> 4;
    const int bm0  = blockIdx.x * BM;
    const int bn0  = blockIdx.y * BN;
    const int nit  = K / BK;

    v8f acc[2][4] = {};

    auto fill = [&](int buf, int k0) {
#pragma unroll
        for (int i = 0; i < 2; ++i) {
            int ci  = tid + i * 256;            // 512 x 16B chunks per tile
            int row = ci >> 2, c16 = ci & 3;    // 4 chunks per 64B row
            async_b128((unsigned)(uintptr_t)&sA[buf][row * 32 + c16 * 8],
                       A  + (size_t)(bm0 + row) * K + k0 + c16 * 8);
            async_b128((unsigned)(uintptr_t)&sB[buf][row * 32 + c16 * 8],
                       Bt + (size_t)(bn0 + row) * K + k0 + c16 * 8);
        }
    };

    fill(0, 0);
    for (int kt = 0; kt < nit; ++kt) {
        const int p = kt & 1;
        wait_async0();
        __syncthreads();
        if (kt + 1 < nit) fill(1 - p, (kt + 1) * BK);

        v16bf afrag[2], bfrag[4];
#pragma unroll
        for (int mi = 0; mi < 2; ++mi) {
            const __bf16* b = &sA[p][(wm * 32 + mi * 16 + l16) * 32 + half * 8];
            afrag[mi] = mk_frag(b, b + 16);
        }
#pragma unroll
        for (int ni = 0; ni < 4; ++ni) {
            const __bf16* b = &sB[p][(wn * 64 + ni * 16 + l16) * 32 + half * 8];
            bfrag[ni] = mk_frag(b, b + 16);
        }
#pragma unroll
        for (int mi = 0; mi < 2; ++mi)
#pragma unroll
            for (int ni = 0; ni < 4; ++ni)
                acc[mi][ni] = __builtin_amdgcn_wmma_f32_16x16x32_bf16(
                    false, afrag[mi], false, bfrag[ni],
                    (short)0, acc[mi][ni], false, false);
        __syncthreads();
    }

#pragma unroll
    for (int mi = 0; mi < 2; ++mi)
#pragma unroll
        for (int ni = 0; ni < 4; ++ni)
#pragma unroll
            for (int vi = 0; vi < 8; ++vi) {
                int r = bm0 + wm * 32 + mi * 16 + vi + 8 * half;
                int c = bn0 + wn * 64 + ni * 16 + l16;
                C[(size_t)r * N + c] = acc[mi][ni][vi];
            }
}

// ---------------------------------------------------------------------------
// RoPE in place on fp32 Q: [S, NH, 128]
// ---------------------------------------------------------------------------
__global__ __launch_bounds__(256)
void rope_q_kernel(float* __restrict__ t, const float* __restrict__ cs,
                   const float* __restrict__ sn) {
    int idx = blockIdx.x * blockDim.x + threadIdx.x;   // S*NH*64
    int j = idx & 63;
    int h = (idx >> 6) % NH;
    int s = idx / (64 * NH);
    float c = cs[s * 64 + j], si = sn[s * 64 + j];
    float* p = t + ((size_t)s * NH + h) * HD + 2 * j;
    float t1 = p[0], t2 = p[1];
    p[0] = t1 * c - t2 * si;
    p[1] = t1 * si + t2 * c;
}

// RoPE on fp32 K, emitting bf16 K: [S, NKV, 128]
__global__ __launch_bounds__(256)
void rope_k_bf16(const float* __restrict__ kin, __bf16* __restrict__ kout,
                 const float* __restrict__ cs, const float* __restrict__ sn) {
    int idx = blockIdx.x * blockDim.x + threadIdx.x;   // S*NKV*64
    int j = idx & 63;
    int h = (idx >> 6) % NKV;
    int s = idx / (64 * NKV);
    float c = cs[s * 64 + j], si = sn[s * 64 + j];
    size_t base = ((size_t)s * NKV + h) * HD + 2 * j;
    float t1 = kin[base], t2 = kin[base + 1];
    v2bf o;
    o[0] = (__bf16)(t1 * c - t2 * si);
    o[1] = (__bf16)(t1 * si + t2 * c);
    *(v2bf*)&kout[base] = o;
}

// ---------------------------------------------------------------------------
// Flash attention. block = (head, 128 q rows); 8 waves x 16 q rows.
// K tiles (bf16 [kv][d]) and V tiles (bf16 [d][kv], pre-transposed in global)
// streamed by async-to-LDS. Output written bf16 for the final projection.
// ---------------------------------------------------------------------------
__global__ __launch_bounds__(256)
void flash_attn(const float* __restrict__ q, const __bf16* __restrict__ k16,
                const __bf16* __restrict__ vt16, __bf16* __restrict__ out) {
    __shared__ __bf16 sK[32 * HD];        // [kv][d]
    __shared__ __bf16 sV[HD * 32];        // [d][kv]
    __shared__ __bf16 sP[8 * 16 * 32];    // per-wave P tiles

    const int tid  = threadIdx.x;
    const int lane = tid & 31;
    const int wave = tid >> 5;
    const int l16  = lane & 15;
    const int half = lane >> 4;
    const int h    = blockIdx.x;
    const int kvh  = h >> 2;              // GQA 4:1
    const int q0   = blockIdx.y * 128 + wave * 16;
    const float scale = 0.08838834764831845f;   // 1/sqrt(128)

    __bf16* sPw = &sP[wave * 512];

    // preload Q fragments (fp32 -> bf16, once per wave)
    v16bf qa[4];
    {
        const float* qrow = q + (size_t)(q0 + l16) * DMODEL + h * HD;
#pragma unroll
        for (int c = 0; c < 4; ++c) {
            Frag16 f;
            int kb = c * 32 + half * 8;
#pragma unroll
            for (int i = 0; i < 8; ++i) {
                f.h[0][i] = (__bf16)qrow[kb + i];
                f.h[1][i] = (__bf16)qrow[kb + 16 + i];
            }
            qa[c] = f.v;
        }
    }

    v8f   o[8] = {};
    float m_r[8], l_r[8];
#pragma unroll
    for (int i = 0; i < 8; ++i) { m_r[i] = -1e30f; l_r[i] = 0.f; }

    const int ntiles = (blockIdx.y + 1) * 4;     // causal bound (uniform/block)
    for (int kvt = 0; kvt < ntiles; ++kvt) {
        const int kv0 = kvt * 32;

        // ---- async stage K (32x128) and V^T (128x32) tiles, 8KB each ----
#pragma unroll
        for (int i = 0; i < 2; ++i) {
            int ci = tid + i * 256;              // 512 chunks of 16B
            int kr = ci >> 4, kc = ci & 15;      // K: 16 chunks per 256B row
            async_b128((unsigned)(uintptr_t)&sK[kr * HD + kc * 8],
                       k16 + (size_t)(kv0 + kr) * KVCOLS + kvh * HD + kc * 8);
            int d = ci >> 2, vc = ci & 3;        // V: 4 chunks per 64B row
            async_b128((unsigned)(uintptr_t)&sV[d * 32 + vc * 8],
                       vt16 + (size_t)(kvh * HD + d) * SEQ + kv0 + vc * 8);
        }
        wait_async0();
        __syncthreads();

        // ---- scores: two 16x16 kv tiles, K=128 chained ----
        v8f s0 = {}, s1 = {};
#pragma unroll
        for (int c = 0; c < 4; ++c) {
            const __bf16* b0 = &sK[(l16)      * HD + c * 32 + half * 8];
            const __bf16* b1 = &sK[(16 + l16) * HD + c * 32 + half * 8];
            v16bf bk0 = mk_frag(b0, b0 + 16);
            v16bf bk1 = mk_frag(b1, b1 + 16);
            s0 = __builtin_amdgcn_wmma_f32_16x16x32_bf16(false, qa[c], false, bk0,
                                                         (short)0, s0, false, false);
            s1 = __builtin_amdgcn_wmma_f32_16x16x32_bf16(false, qa[c], false, bk1,
                                                         (short)0, s1, false, false);
        }

        // ---- causal mask + online softmax (row = vi + 8*half, col = l16) ----
#pragma unroll
        for (int vi = 0; vi < 8; ++vi) {
            int rq = q0 + vi + 8 * half;
            float x0 = (kv0 + l16      <= rq) ? s0[vi] * scale : -1e9f;
            float x1 = (kv0 + 16 + l16 <= rq) ? s1[vi] * scale : -1e9f;
            float mloc = fmaxf(x0, x1);
#pragma unroll
            for (int off = 1; off < 16; off <<= 1)
                mloc = fmaxf(mloc, __shfl_xor(mloc, off, 32));
            float mnew = fmaxf(m_r[vi], mloc);
            float al   = __expf(m_r[vi] - mnew);
            float p0   = __expf(x0 - mnew);
            float p1   = __expf(x1 - mnew);
            float ps   = p0 + p1;
#pragma unroll
            for (int off = 1; off < 16; off <<= 1)
                ps += __shfl_xor(ps, off, 32);
            l_r[vi] = l_r[vi] * al + ps;
            m_r[vi] = mnew;
#pragma unroll
            for (int t = 0; t < 8; ++t) o[t][vi] *= al;
            sPw[(vi + 8 * half) * 32 + l16]      = (__bf16)p0;
            sPw[(vi + 8 * half) * 32 + 16 + l16] = (__bf16)p1;
        }
        asm volatile("s_wait_dscnt 0x0" ::: "memory");   // wave-local P ready

        // ---- P(16x32) @ V(32x128): 8 WMMAs ----
        {
            const __bf16* pa = &sPw[l16 * 32 + half * 8];
            v16bf ap = mk_frag(pa, pa + 16);
#pragma unroll
            for (int t = 0; t < 8; ++t) {
                const __bf16* bv = &sV[(t * 16 + l16) * 32 + half * 8];
                v16bf bfv = mk_frag(bv, bv + 16);
                o[t] = __builtin_amdgcn_wmma_f32_16x16x32_bf16(false, ap, false, bfv,
                                                               (short)0, o[t], false, false);
            }
        }
        __syncthreads();
    }

    // ---- normalize + store bf16 [S, NH*HD] ----
#pragma unroll
    for (int t = 0; t < 8; ++t)
#pragma unroll
        for (int vi = 0; vi < 8; ++vi) {
            int rg = q0 + vi + 8 * half;
            out[(size_t)rg * DMODEL + h * HD + t * 16 + l16] =
                (__bf16)(o[t][vi] / l_r[vi]);
        }
}

// ---------------------------------------------------------------------------
extern "C" void kernel_launch(void* const* d_in, const int* in_sizes, int n_in,
                              void* d_out, int out_size, void* d_ws, size_t ws_size,
                              hipStream_t stream) {
    const float* x  = (const float*)d_in[0];
    const float* wq = (const float*)d_in[1];
    const float* wk = (const float*)d_in[2];
    const float* wv = (const float*)d_in[3];
    const float* wo = (const float*)d_in[4];
    const float* cs = (const float*)d_in[5];
    const float* sn = (const float*)d_in[6];
    // d_in[7] additive mask: causal mask applied in-register.

    // fp32 scratch
    float* qb = (float*)d_ws;                       // [S, 4096]
    float* kb = qb + (size_t)SEQ * DMODEL;          // [S, 1024]
    float* vb = kb + (size_t)SEQ * KVCOLS;          // [S, 1024]
    // bf16 scratch
    __bf16* x16  = (__bf16*)(vb + (size_t)SEQ * KVCOLS);
    __bf16* wqT  = x16  + (size_t)SEQ * DMODEL;     // [4096, 4096]
    __bf16* wkT  = wqT  + (size_t)DMODEL * DMODEL;  // [1024, 4096]
    __bf16* wvT  = wkT  + (size_t)KVCOLS * DMODEL;  // [1024, 4096]
    __bf16* woT  = wvT  + (size_t)KVCOLS * DMODEL;  // [4096, 4096]
    __bf16* kb16 = woT  + (size_t)DMODEL * DMODEL;  // [S, 1024]
    __bf16* vt16 = kb16 + (size_t)SEQ * KVCOLS;     // [1024, S]
    __bf16* ab16 = vt16 + (size_t)SEQ * KVCOLS;     // [S, 4096]

    dim3 blk(256);

    // one-time bf16 conversions / transposes
    cvt_f32_bf16<<<(SEQ * DMODEL) / 1024, blk, 0, stream>>>(x, x16);
    transpose_cvt<<<dim3(DMODEL / 32, DMODEL / 32), blk, 0, stream>>>(wq, wqT, DMODEL, DMODEL);
    transpose_cvt<<<dim3(KVCOLS / 32, DMODEL / 32), blk, 0, stream>>>(wk, wkT, DMODEL, KVCOLS);
    transpose_cvt<<<dim3(KVCOLS / 32, DMODEL / 32), blk, 0, stream>>>(wv, wvT, DMODEL, KVCOLS);
    transpose_cvt<<<dim3(DMODEL / 32, DMODEL / 32), blk, 0, stream>>>(wo, woT, DMODEL, DMODEL);

    // projections
    gemm_bf16_wmma<<<dim3(SEQ / BM, DMODEL / BN), blk, 0, stream>>>(x16, wqT, qb, SEQ, DMODEL, DMODEL);
    gemm_bf16_wmma<<<dim3(SEQ / BM, KVCOLS / BN), blk, 0, stream>>>(x16, wkT, kb, SEQ, KVCOLS, DMODEL);
    gemm_bf16_wmma<<<dim3(SEQ / BM, KVCOLS / BN), blk, 0, stream>>>(x16, wvT, vb, SEQ, KVCOLS, DMODEL);

    // RoPE (+ K to bf16), V^T to bf16
    rope_q_kernel<<<(SEQ * NH  * 64) / 256, blk, 0, stream>>>(qb, cs, sn);
    rope_k_bf16  <<<(SEQ * NKV * 64) / 256, blk, 0, stream>>>(kb, kb16, cs, sn);
    transpose_cvt<<<dim3(KVCOLS / 32, SEQ / 32), blk, 0, stream>>>(vb, vt16, SEQ, KVCOLS);

    // attention -> bf16 activations
    flash_attn<<<dim3(NH, SEQ / 128), blk, 0, stream>>>(qb, kb16, vt16, ab16);

    // output projection -> fp32 result
    gemm_bf16_wmma<<<dim3(SEQ / BM, DMODEL / BN), blk, 0, stream>>>(ab16, woT, (float*)d_out, SEQ, DMODEL, DMODEL);
}
